// DN_21758304321889
// MI455X (gfx1250) — matrix operand order repaired and checked by reading, not compiled, not run on hardware
//
#include <hip/hip_runtime.h>

typedef __attribute__((ext_vector_type(16))) _Float16 v16h;
typedef __attribute__((ext_vector_type(8)))  _Float16 v8h;
typedef __attribute__((ext_vector_type(8)))  float    v8f;
typedef __attribute__((ext_vector_type(4)))  float    v4f;

#define KDIM 4096
#define YDIM 8192
#define BDIM 256
#define ZDIM 101
#define BM   64
#define BN   128
#define BK   32
#define LDA  40   // padded halfs per LDS row (80B: 16B-aligned, spreads banks)
#define LDB  40
#define TOPK 32
#define GAPV 0.01f

// ---------------- row inverse L2 norms for x (256 rows) and x2y_w (8192 rows)
__global__ __launch_bounds__(256) void dn_norms(const float* __restrict__ x,
                                                const float* __restrict__ w,
                                                float* __restrict__ invx,
                                                float* __restrict__ invw) {
  __shared__ float red[256];
  const int t = threadIdx.x;
  const int rb = blockIdx.x;
  const float* src;
  float* dst;
  if (rb < BDIM) { src = x + (size_t)rb * KDIM; dst = invx + rb; }
  else           { int r = rb - BDIM; src = w + (size_t)r * KDIM; dst = invw + r; }
  float s = 0.f;
  const v4f* p = (const v4f*)src;
#pragma unroll 4
  for (int i = t; i < KDIM / 4; i += 256) {
    v4f v = p[i];
    s += v[0]*v[0] + v[1]*v[1] + v[2]*v[2] + v[3]*v[3];
  }
  red[t] = s;
  __syncthreads();
  for (int off = 128; off > 0; off >>= 1) {
    if (t < off) red[t] += red[t + off];
    __syncthreads();
  }
  if (t == 0) {
    float n = sqrtf(red[0]);
    *dst = 1.0f / fmaxf(n, 1e-12f);
  }
}

// ---------------- per-neuron class = argmax over Z of y2z_w[z, y] (first max)
__global__ __launch_bounds__(256) void dn_cls(const float* __restrict__ y2z,
                                              int* __restrict__ cls) {
  int y = blockIdx.x * 256 + threadIdx.x;
  if (y >= YDIM) return;
  float best = y2z[y];
  int bi = 0;
#pragma unroll 4
  for (int z = 1; z < ZDIM; ++z) {
    float v = y2z[(size_t)z * YDIM + y];
    if (v > best) { best = v; bi = z; }
  }
  cls[y] = bi;
}

// ---------------- main WMMA GEMM: resp[b,y] = (x[b,:].w[y,:]) * invx[b]*invw[y]*act[y]
// Software-pipelined, double-buffered LDS, one barrier per K-step.
__global__ __launch_bounds__(256) void dn_gemm(const float* __restrict__ x,
                                               const float* __restrict__ w,
                                               const float* __restrict__ invx,
                                               const float* __restrict__ invw,
                                               const float* __restrict__ age,
                                               float* __restrict__ resp) {
  __shared__ __align__(16) _Float16 As[2][BM * LDA];
  __shared__ __align__(16) _Float16 Bs[2][BN * LDB];

  const int tid  = threadIdx.x;
  const int lane = tid & 31;
  const int wave = tid >> 5;    // 0..7
  const int wm   = wave >> 2;   // 0..1  (M sub-block)
  const int wn   = wave & 3;    // 0..3  (N sub-block)
  const int bm   = blockIdx.y * BM;
  const int bn   = blockIdx.x * BN;

  v8f acc[2][2] = {};

  // A staging: thread loads row = tid>>2 (0..63), kk = (tid&3)*8  -> 8 f32
  const int a_row = tid >> 2;
  const int a_kk  = (tid & 3) * 8;
  // B staging: thread loads row = tid>>1 (0..127), kk = (tid&1)*16 -> 16 f32
  const int b_row = tid >> 1;
  const int b_kk  = (tid & 1) * 16;

  const float* aptr = x + (size_t)(bm + a_row) * KDIM + a_kk;
  const float* bptr = w + (size_t)(bn + b_row) * KDIM + b_kk;

  const int laneoff = (lane >> 4) * 8;   // A fragment K sub-offset
  const int koffB   = (lane >> 4) * 8;   // B fragment K sub-offset
  const int lrow    = lane & 15;

  // convert + store one staged tile (held in regs) into LDS buffer `bsel`
  auto stage = [&](int bsel, v4f a0, v4f a1, v4f q0, v4f q1, v4f q2, v4f q3) {
    v8h h;
#pragma unroll
    for (int i = 0; i < 4; ++i) { h[i] = (_Float16)a0[i]; h[i + 4] = (_Float16)a1[i]; }
    *(v8h*)(&As[bsel][a_row * LDA + a_kk]) = h;
    v8h h0, h1;
#pragma unroll
    for (int i = 0; i < 4; ++i) {
      h0[i] = (_Float16)q0[i]; h0[i + 4] = (_Float16)q1[i];
      h1[i] = (_Float16)q2[i]; h1[i + 4] = (_Float16)q3[i];
    }
    *(v8h*)(&Bs[bsel][b_row * LDB + b_kk])     = h0;
    *(v8h*)(&Bs[bsel][b_row * LDB + b_kk + 8]) = h1;
  };

  // ---- prologue: tile 0 -> LDS buf0; tile 1 -> regs
  v4f ra0 = *(const v4f*)(aptr);
  v4f ra1 = *(const v4f*)(aptr + 4);
  v4f rb0 = *(const v4f*)(bptr);
  v4f rb1 = *(const v4f*)(bptr + 4);
  v4f rb2 = *(const v4f*)(bptr + 8);
  v4f rb3 = *(const v4f*)(bptr + 12);
  stage(0, ra0, ra1, rb0, rb1, rb2, rb3);
  aptr += BK; bptr += BK;
  ra0 = *(const v4f*)(aptr);
  ra1 = *(const v4f*)(aptr + 4);
  rb0 = *(const v4f*)(bptr);
  rb1 = *(const v4f*)(bptr + 4);
  rb2 = *(const v4f*)(bptr + 8);
  rb3 = *(const v4f*)(bptr + 12);
  __syncthreads();

  const int NT = KDIM / BK;  // 128
  int buf = 0;
  for (int kt = 0; kt < NT; ++kt) {
    // 1. issue global loads for tile kt+2 (latency hidden behind compute)
    v4f na0 = {}, na1 = {}, nb0 = {}, nb1 = {}, nb2 = {}, nb3 = {};
    if (kt + 2 < NT) {
      aptr += BK; bptr += BK;
      __builtin_prefetch(aptr + BK, 0, 0);  // global_prefetch_b8 one more tile out
      __builtin_prefetch(bptr + BK, 0, 0);
      na0 = *(const v4f*)(aptr);
      na1 = *(const v4f*)(aptr + 4);
      nb0 = *(const v4f*)(bptr);
      nb1 = *(const v4f*)(bptr + 4);
      nb2 = *(const v4f*)(bptr + 8);
      nb3 = *(const v4f*)(bptr + 12);
    }

    // 2. compute tile kt from LDS buf (documented CDNA5 16-bit WMMA layouts)
    v16h Af[2], Bf[2];
#pragma unroll
    for (int tm = 0; tm < 2; ++tm) {
      int row = wm * 32 + tm * 16 + lrow;
      v8h x0 = *(const v8h*)(&As[buf][row * LDA + laneoff]);       // K = laneoff+0..7
      v8h x1 = *(const v8h*)(&As[buf][row * LDA + 16 + laneoff]);  // K = 16+laneoff+0..7
      Af[tm] = __builtin_shufflevector(x0, x1, 0,1,2,3,4,5,6,7,8,9,10,11,12,13,14,15);
    }
#pragma unroll
    for (int tn = 0; tn < 2; ++tn) {
      int col = wn * 32 + tn * 16 + lrow;
      v8h y0 = *(const v8h*)(&Bs[buf][col * LDB + koffB]);         // K = koffB+0..7
      v8h y1 = *(const v8h*)(&Bs[buf][col * LDB + 16 + koffB]);    // K = 16+koffB+0..7
      Bf[tn] = __builtin_shufflevector(y0, y1, 0,1,2,3,4,5,6,7,8,9,10,11,12,13,14,15);
    }
#pragma unroll
    for (int tm = 0; tm < 2; ++tm)
#pragma unroll
      for (int tn = 0; tn < 2; ++tn)
        acc[tm][tn] = __builtin_amdgcn_wmma_f32_16x16x32_f16(
            false, Af[tm], false, Bf[tn], (short)0, acc[tm][tn], false, false);

    // 3. convert + store tile kt+1 into the other buffer
    if (kt + 1 < NT) stage(buf ^ 1, ra0, ra1, rb0, rb1, rb2, rb3);
    if (kt + 2 < NT) {
      ra0 = na0; ra1 = na1;
      rb0 = nb0; rb1 = nb1; rb2 = nb2; rb3 = nb3;
    }
    __syncthreads();
    buf ^= 1;
  }

  // ---- epilogue: scale by invx[b]*invw[y]*act[y], scatter to resp
#pragma unroll
  for (int tn = 0; tn < 2; ++tn) {
    int gn = bn + wn * 32 + tn * 16 + lrow;
    float sn = invw[gn] * (age[gn] >= 1.0f ? 1.0f : 0.0f);
#pragma unroll
    for (int tm = 0; tm < 2; ++tm) {
      int gm0 = bm + wm * 32 + tm * 16 + ((lane >> 4) << 3);
#pragma unroll
      for (int r = 0; r < 8; ++r) {
        int gm = gm0 + r;
        resp[(size_t)gm * YDIM + gn] = acc[tm][tn][r] * invx[gm] * sn;
      }
    }
  }
}

// ---------------- per-row top-K extraction + decision + output gather
__global__ __launch_bounds__(256) void dn_select(const float* __restrict__ resp,
                                                 const int* __restrict__ cls,
                                                 const int* __restrict__ zin,
                                                 const float* __restrict__ y2z,
                                                 float* __restrict__ out) {
  __shared__ __align__(16) float vals[YDIM];
  __shared__ float rv[256];
  __shared__ int   ri[256];
  __shared__ float tv[TOPK];
  __shared__ int   ti[TOPK];
  __shared__ int   s_maxidx;

  const int b = blockIdx.x;
  const int t = threadIdx.x;
  const float* row = resp + (size_t)b * YDIM;

  // Stage the 32KB row into LDS with async copies (ASYNCcnt path, no VGPR trip):
  // each of 256 threads issues 8x GLOBAL_LOAD_ASYNC_TO_LDS_B128 (16B/lane).
#pragma unroll
  for (int i = t * 4; i < YDIM; i += 256 * 4) {
    unsigned int laddr = (unsigned int)(uintptr_t)(&vals[i]);
    const float* gaddr = row + i;
    asm volatile("global_load_async_to_lds_b128 %0, %1, off"
                 :: "v"(laddr), "v"(gaddr)
                 : "memory");
  }
  asm volatile("s_wait_asynccnt 0" ::: "memory");
  __syncthreads();

  for (int k = 0; k < TOPK; ++k) {
    float bv = -__builtin_inff();
    int   bi = 0x7fffffff;
#pragma unroll 4
    for (int i = t; i < YDIM; i += 256) {
      float v = vals[i];
      if (v > bv || (v == bv && i < bi)) { bv = v; bi = i; }
    }
    rv[t] = bv; ri[t] = bi;
    __syncthreads();
    for (int off = 128; off > 0; off >>= 1) {
      if (t < off) {
        float ov = rv[t + off]; int oi = ri[t + off];
        if (ov > rv[t] || (ov == rv[t] && oi < ri[t])) { rv[t] = ov; ri[t] = oi; }
      }
      __syncthreads();
    }
    if (t == 0) {
      tv[k] = rv[0]; ti[k] = ri[0];
      vals[ri[0]] = -__builtin_inff();
    }
    __syncthreads();
  }

  if (t == 0) {
    const int zb = zin[b] + 1;
    int maxidx = ti[0];
    const bool r0nz   = (tv[0] != 0.0f);
    const bool active = (cls[ti[0]] == 0) && r0nz;
    int first = -1;
    for (int j = 1; j < TOPK; ++j) {
      if (cls[ti[j]] != 0 || tv[j] == 0.0f) { first = j; break; }
    }
    if (active && first > 0) {
      int   fc = cls[ti[first]];
      float fr = tv[first];
      if (fc != 0 && fr != 0.0f) maxidx = ti[first];
    }
    const bool pass2  = r0nz && (cls[maxidx] != zb);
    const bool gap_ok = (tv[0] - tv[1]) < GAPV;
    const bool cand1  = pass2 && (tv[1] != 0.0f) && (cls[ti[1]] == zb);
    if (cand1 && gap_ok) maxidx = ti[1];
    const bool remaining = pass2 && !cand1;
    const bool cand2 = remaining && (tv[2] != 0.0f) && (cls[ti[2]] == zb);
    if (cand2 && gap_ok) maxidx = ti[2];
    s_maxidx = maxidx;
  }
  __syncthreads();
  const int mi = s_maxidx;
  if (t < ZDIM) out[(size_t)b * ZDIM + t] = y2z[(size_t)t * YDIM + mi];
}

extern "C" void kernel_launch(void* const* d_in, const int* in_sizes, int n_in,
                              void* d_out, int out_size, void* d_ws, size_t ws_size,
                              hipStream_t stream) {
  const float* x    = (const float*)d_in[0];   // (256, 64, 64) f32
  const int*   z    = (const int*)d_in[1];     // (256,) i32
  const float* x2y  = (const float*)d_in[2];   // (8192, 4096) f32
  const float* y2z  = (const float*)d_in[3];   // (101, 8192) f32
  const float* age  = (const float*)d_in[4];   // (1, 8192) f32
  float* out = (float*)d_out;                  // (256, 101) f32

  char* ws = (char*)d_ws;
  float* resp = (float*)ws;                                  // 256*8192 f32 = 8 MB
  float* invx = (float*)(ws + (size_t)BDIM * YDIM * 4);      // 256 f32
  float* invw = invx + BDIM;                                 // 8192 f32
  int*   cls  = (int*)(invw + YDIM);                         // 8192 i32

  dn_norms<<<BDIM + YDIM, 256, 0, stream>>>(x, x2y, invx, invw);
  dn_cls<<<YDIM / 256, 256, 0, stream>>>(y2z, cls);
  dim3 grid(YDIM / BN, BDIM / BM);  // (64, 4)
  dn_gemm<<<grid, 256, 0, stream>>>(x, x2y, invx, invw, age, resp);
  dn_select<<<BDIM, 256, 0, stream>>>(resp, cls, z, y2z, out);
}
